// SoftProposal_41970420416737
// MI455X (gfx1250) — matrix operand order repaired and checked by reading, not compiled, not run on hardware
//
#include <hip/hip_runtime.h>
#include <hip/hip_bf16.h>
#include <math.h>

// Problem constants (from reference: x = [16, 1024, 32, 32] fp32)
#define BATCH 16
#define CH    1024
#define NPOS  1024          // 32*32
#define NN    (NPOS * NPOS) // 1 << 20
#define FACTOR 153.6f       // 0.15 * N
#define EPSV  1e-12f
#define TOLV  1e-4f

typedef __attribute__((ext_vector_type(16))) _Float16 v16h;
typedef __attribute__((ext_vector_type(8)))  float    v8f;

// ---------------------------------------------------------------------------
// Kernel 1: G[b] = X^T X   (X = x[b] as [C][N], f16 WMMA, f32 accumulate)
// 128x128 output tile per block, 256 threads = 8 waves, wave w owns a 16x128
// row strip (8 accumulators). LDS tiles are stored transposed [n][c] with a
// 40-half row stride (80B, 16B-aligned) so WMMA fragments are contiguous
// row reads matching the CDNA5 16-bit A/B VGPR layouts.
// ---------------------------------------------------------------------------
__global__ __launch_bounds__(256) void sp_gemm(const float* __restrict__ x,
                                               float* __restrict__ G) {
  const int b  = blockIdx.z;
  const int i0 = blockIdx.y * 128;
  const int j0 = blockIdx.x * 128;
  const int tid  = threadIdx.x;
  const int wave = tid >> 5;
  const int lane = tid & 31;

  __shared__ _Float16 tA[128][40];
  __shared__ _Float16 tB[128][40];

  const float* xb = x + (size_t)b * CH * NPOS;

  v8f acc[8];
#pragma unroll
  for (int j = 0; j < 8; ++j)
#pragma unroll
    for (int v = 0; v < 8; ++v) acc[j][v] = 0.0f;

  for (int kc = 0; kc < CH; kc += 32) {
    // stage (convert f32 -> f16, transpose into [n][c])
#pragma unroll
    for (int e = tid; e < 128 * 32; e += 256) {
      const int n = e & 127;      // consecutive tid -> consecutive n (coalesced)
      const int c = e >> 7;       // 0..31
      const size_t row = (size_t)(kc + c) * NPOS;
      tA[n][c] = (_Float16)xb[row + (i0 + n)];
      tB[n][c] = (_Float16)xb[row + (j0 + n)];
    }
    __syncthreads();

    // A fragment: 16x32, rows i0 + wave*16 .. +15
    // lane<16: M=lane, K{0..7,16..23}; lane>=16: M=lane-16, K{8..15,24..31}
    {
      const int m  = lane & 15;
      const int hk = (lane >> 4) * 8;
      const _Float16* pr = &tA[wave * 16 + m][0];
      v16h a;
#pragma unroll
      for (int e = 0; e < 8; ++e) { a[e] = pr[hk + e]; a[8 + e] = pr[16 + hk + e]; }

#pragma unroll
      for (int j = 0; j < 8; ++j) {
        // B fragment: 32x16. lane<16: N=lane, K{0..7,16..23}; lane>=16: N=lane-16, K{8..15,24..31}
        const _Float16* pc = &tB[j * 16 + (lane & 15)][(lane >> 4) * 8];
        v16h bf;
#pragma unroll
        for (int e = 0; e < 16; ++e) bf[e] = pc[e < 8 ? e : e + 8];
        acc[j] = __builtin_amdgcn_wmma_f32_16x16x32_f16(
            false, a, false, bf, (short)0, acc[j], false, false);
      }
    }
    __syncthreads();
  }

  // C/D layout: lane<16 -> M = v, N = lane; lane>=16 -> M = 8+v, N = lane-16
  float* Gb = G + (size_t)b * NN;
  const int ncol = lane & 15;
  const int mhi  = (lane >> 4) * 8;
#pragma unroll
  for (int j = 0; j < 8; ++j)
#pragma unroll
    for (int v = 0; v < 8; ++v)
      Gb[(size_t)(i0 + wave * 16 + mhi + v) * NPOS + (j0 + j * 16 + ncol)] = acc[j][v];
}

// ---------------------------------------------------------------------------
// Kernel 2: sq[b][n] = G[b][n][n]   (extract diag before in-place overwrite)
// ---------------------------------------------------------------------------
__global__ void sp_diag(const float* __restrict__ G, float* __restrict__ sq) {
  const int b = blockIdx.y;
  const int n = blockIdx.x * 256 + threadIdx.x;
  sq[b * NPOS + n] = G[(size_t)b * NN + (size_t)n * NPOS + n];
}

// ---------------------------------------------------------------------------
// Kernel 3: in-place  G <- U = dm(i,j) * sqrt(max(sq_i + sq_j - 2 G, 0))
// ---------------------------------------------------------------------------
__global__ void sp_u(float* __restrict__ G, const float* __restrict__ sq) {
  const size_t flat = (size_t)blockIdx.x * 256 + threadIdx.x;  // < B*NN
  const int b = (int)(flat >> 20);
  const int r = (int)(flat & (NN - 1));
  const int i = r >> 10;
  const int j = r & (NPOS - 1);
  const float g  = G[flat];
  const float d2 = sq[b * NPOS + i] + sq[b * NPOS + j] - 2.0f * g;
  const float fd = sqrtf(fmaxf(d2, 0.0f));
  const int dy = (i >> 5) - (j >> 5);
  const int dx = (i & 31) - (j & 31);
  const float dm = __expf(-(float)(dy * dy + dx * dx) / FACTOR);
  G[flat] = dm * fd;
}

// ---------------------------------------------------------------------------
// Kernel 4: colsum[b][j] = sum_i U[b][i][j]
// ---------------------------------------------------------------------------
__global__ __launch_bounds__(64) void sp_colsum(const float* __restrict__ U,
                                                float* __restrict__ colsum) {
  const int b = blockIdx.y;
  const int j = blockIdx.x * 64 + threadIdx.x;
  const float* Ub = U + (size_t)b * NN;
  float s = 0.0f;
  for (int i = 0; i < NPOS; ++i) s += Ub[(size_t)i * NPOS + j];
  colsum[b * NPOS + j] = s;
}

// ---------------------------------------------------------------------------
// Kernel 5: p = 1/N ; q = p / max(colsum, eps) ; done = 0
// ---------------------------------------------------------------------------
__global__ void sp_pinit(const float* __restrict__ colsum, float* __restrict__ p,
                         float* __restrict__ q, int* __restrict__ done) {
  const int b = blockIdx.y;
  const int n = blockIdx.x * 256 + threadIdx.x;
  const float p0 = 1.0f / (float)NPOS;
  p[b * NPOS + n] = p0;
  q[b * NPOS + n] = p0 / fmaxf(colsum[b * NPOS + n], EPSV);
  if (blockIdx.x == 0 && threadIdx.x == 0) done[b] = 0;
}

// ---------------------------------------------------------------------------
// Kernel 6a: pnew_raw = U @ q   (T@p with column-norm folded into q)
// block = 16 rows of one batch; 16 threads per row.
// ---------------------------------------------------------------------------
__global__ __launch_bounds__(256) void sp_matvec(const float* __restrict__ U,
                                                 const float* __restrict__ q,
                                                 float* __restrict__ pnew,
                                                 const int* __restrict__ done) {
  const int b = blockIdx.y;
  if (done[b]) return;
  const int row = blockIdx.x * 16 + (threadIdx.x >> 4);
  const int jo  = threadIdx.x & 15;
  const float* Ub = U + (size_t)b * NN + (size_t)row * NPOS;
  const float* qb = q + b * NPOS;
  float s = 0.0f;
  for (int j = jo; j < NPOS; j += 16) s += Ub[j] * qb[j];
  __shared__ float red[256];
  red[threadIdx.x] = s;
  __syncthreads();
  if (jo == 0) {
    float tot = 0.0f;
#pragma unroll
    for (int e = 0; e < 16; ++e) tot += red[threadIdx.x + e];
    pnew[b * NPOS + row] = tot;
  }
}

// ---------------------------------------------------------------------------
// Kernel 6b: normalize, L1 diff, freeze flag, refresh p and q (one block/batch)
// ---------------------------------------------------------------------------
__global__ __launch_bounds__(256) void sp_update(const float* __restrict__ pnew,
                                                 const float* __restrict__ colsum,
                                                 float* __restrict__ p,
                                                 float* __restrict__ q,
                                                 int* __restrict__ done) {
  const int b = blockIdx.x;
  if (done[b]) return;
  const int t = threadIdx.x;
  const float* pnb = pnew + b * NPOS;
  const float* cb  = colsum + b * NPOS;
  float* pb = p + b * NPOS;
  float* qb = q + b * NPOS;

  __shared__ float red[256];
  float vals[4];
  float s = 0.0f;
#pragma unroll
  for (int k = 0; k < 4; ++k) { vals[k] = pnb[t + 256 * k]; s += vals[k]; }
  red[t] = s;
  __syncthreads();
  for (int off = 128; off > 0; off >>= 1) {
    if (t < off) red[t] += red[t + off];
    __syncthreads();
  }
  const float inv = 1.0f / fmaxf(red[0], EPSV);
  __syncthreads();

  float newv[4];
  float dpart = 0.0f;
#pragma unroll
  for (int k = 0; k < 4; ++k) {
    const int n = t + 256 * k;
    newv[k] = vals[k] * inv;
    dpart += fabsf(newv[k] - pb[n]);
  }
  red[t] = dpart;
  __syncthreads();
  for (int off = 128; off > 0; off >>= 1) {
    if (t < off) red[t] += red[t + off];
    __syncthreads();
  }
  const float diff = red[0];

#pragma unroll
  for (int k = 0; k < 4; ++k) {
    const int n = t + 256 * k;
    pb[n] = newv[k];
    qb[n] = newv[k] / fmaxf(cb[n], EPSV);
  }
  if (t == 0 && diff < TOLV) done[b] = 1;
}

// ---------------------------------------------------------------------------
// Kernel 7: out = x * p (broadcast over channels), float4 vectorized
// ---------------------------------------------------------------------------
__global__ void sp_scale(const float* __restrict__ x, const float* __restrict__ p,
                         float* __restrict__ out) {
  const size_t idx4 = ((size_t)blockIdx.x * 256 + threadIdx.x) * 4;
  const int b = (int)(idx4 >> 20);  // C*N = 1<<20
  const int n = (int)(idx4 & (NPOS - 1));
  const float4 xv = *(const float4*)(x + idx4);
  const float* pb = p + b * NPOS;
  float4 ov;
  ov.x = xv.x * pb[n + 0];
  ov.y = xv.y * pb[n + 1];
  ov.z = xv.z * pb[n + 2];
  ov.w = xv.w * pb[n + 3];
  *(float4*)(out + idx4) = ov;
}

// ---------------------------------------------------------------------------
extern "C" void kernel_launch(void* const* d_in, const int* in_sizes, int n_in,
                              void* d_out, int out_size, void* d_ws, size_t ws_size,
                              hipStream_t stream) {
  (void)in_sizes; (void)n_in; (void)out_size; (void)ws_size;
  const float* x = (const float*)d_in[0];
  float* out = (float*)d_out;

  // workspace layout (bytes)
  char* ws = (char*)d_ws;
  float* G      = (float*)(ws);                               // B*NN   = 64 MB
  float* sq     = (float*)(ws + (size_t)BATCH * NN * 4);      // B*N
  float* colsum = sq + BATCH * NPOS;
  float* p      = colsum + BATCH * NPOS;
  float* q      = p + BATCH * NPOS;
  float* pnew   = q + BATCH * NPOS;
  int*   done   = (int*)(pnew + BATCH * NPOS);

  // 1) Gram matrices via WMMA
  sp_gemm<<<dim3(8, 8, BATCH), 256, 0, stream>>>(x, G);
  // 2) diag -> sq
  sp_diag<<<dim3(NPOS / 256, BATCH), 256, 0, stream>>>(G, sq);
  // 3) G <- U (in place)
  sp_u<<<(unsigned)(((size_t)BATCH * NN) / 256), 256, 0, stream>>>(G, sq);
  // 4) column sums
  sp_colsum<<<dim3(NPOS / 64, BATCH), 64, 0, stream>>>(G, colsum);
  // 5) init p, q, done
  sp_pinit<<<dim3(NPOS / 256, BATCH), 256, 0, stream>>>(colsum, p, q, done);
  // 6) 20 power-iteration steps with freeze semantics
  for (int it = 0; it < 20; ++it) {
    sp_matvec<<<dim3(NPOS / 16, BATCH), 256, 0, stream>>>(G, q, pnew, done);
    sp_update<<<BATCH, 256, 0, stream>>>(pnew, colsum, p, q, done);
  }
  // 7) out = x * p
  sp_scale<<<(unsigned)(((size_t)BATCH * CH * NPOS) / 4 / 256), 256, 0, stream>>>(x, p, out);
}
